// MemorizingTransformer_88545045774777
// MI455X (gfx1250) — compile-verified, matrix operand order checked by reading, not run on hardware
//
#include <hip/hip_runtime.h>
#include <hip/hip_bf16.h>
#include <math.h>

// Model constants (from reference)
#define Bb      2
#define Nn      1024
#define DIMd    512
#define HEADSh  8
#define DHd     64
#define Vv      32000
#define Mm      2048
#define KTOP    32
#define SCALEf  0.125f          // DH^-0.5
#define SQRTDH  8.0f            // DH^0.5

typedef __attribute__((ext_vector_type(16))) _Float16 v16h;
typedef __attribute__((ext_vector_type(8)))  float    v8f;

// ---------------------------------------------------------------------------
// Tensor Data Mover support (gfx1250). Arity differs between toolchains:
// ROCm 7.2 clang-22: 5 args; therock clang-23 (ships the TDM header): 6 args.
// ---------------------------------------------------------------------------
#if defined(__has_builtin)
#  if __has_builtin(__builtin_amdgcn_tensor_load_to_lds)
#    define HAVE_TDM 1
#  endif
#endif
#ifndef HAVE_TDM
#  define HAVE_TDM 0
#endif
#if __has_include(<hip/amd_detail/amd_gfx1250_TDM.h>)
#  define TDM_ARITY6 1
#else
#  define TDM_ARITY6 0
#endif

#if HAVE_TDM
typedef unsigned int u32x4 __attribute__((ext_vector_type(4)));
typedef int          i32x8 __attribute__((ext_vector_type(8)));
typedef int          i32x4 __attribute__((ext_vector_type(4)));

__device__ __forceinline__ unsigned lds_off32(const void* p) {
    // generic pointer to LDS: low 32 bits are the wave-relative LDS offset
    return (unsigned)(unsigned long long)(uintptr_t)p;
}

// 2D tile DMA: global (row-major, stride0 elements of 4B) -> LDS, optional
// row padding (pad_interval code per ISA: 4 -> every 32 dwords, 6 -> every 128;
// pad_amount 0 -> 1 dword).
__device__ __forceinline__ void tdm_load_2d(unsigned lds_addr, const void* gptr,
                                            unsigned tile_d0, unsigned tile_d1,
                                            unsigned long long stride0,
                                            unsigned pad_code, int pad_en)
{
    unsigned long long ga = (unsigned long long)(uintptr_t)gptr;
    u32x4 g0;
    g0.x = 1u;                                    // count=1, user desc, no gather
    g0.y = lds_addr;                              // D#.lds_addr
    g0.z = (unsigned)(ga & 0xffffffffull);        // global_addr[31:0]
    g0.w = (unsigned)((ga >> 32) & 0x01fffffful)  // global_addr[56:32]
         | (2u << 30);                            // type=2 (image)
    i32x8 g1;
    g1[0] = (int)((2u << 16)                      // data_size=4B
          | (pad_en ? (1u << 20) : 0u)            // pad_enable
          | ((pad_code & 7u) << 22));             // pad_interval (amount=0 -> 1 dword)
    g1[1] = (int)((tile_d0 & 0xffffu) << 16);     // tensor_dim0[15:0] (= tile, in-bounds)
    g1[2] = (int)((tile_d0 >> 16) | ((tile_d1 & 0xffffu) << 16)); // dim0 hi | dim1 lo
    g1[3] = (int)((tile_d1 >> 16) | ((tile_d0 & 0xffffu) << 16)); // dim1 hi | tile_dim0
    g1[4] = (int)(tile_d1 & 0xffffu);             // tile_dim1 (tile_dim2 = 0)
    g1[5] = (int)(stride0 & 0xffffffffull);       // tensor_dim0_stride[31:0]
    g1[6] = (int)((stride0 >> 32) & 0xffffull);   // stride hi | dim1_stride lo (0)
    g1[7] = 0;
    i32x4 gz = {0, 0, 0, 0};
#if TDM_ARITY6
    i32x8 gz8 = {0, 0, 0, 0, 0, 0, 0, 0};
    __builtin_amdgcn_tensor_load_to_lds(g0, g1, gz, gz, gz8, 0);
#else
    __builtin_amdgcn_tensor_load_to_lds(g0, g1, gz, gz, 0);
#endif
}
#endif // HAVE_TDM

// ---------------------------------------------------------------------------
// Embedding gather
// ---------------------------------------------------------------------------
__global__ void embed_k(const int* __restrict__ tokens,
                        const float* __restrict__ tok_emb,
                        float* __restrict__ x)
{
    const int row = blockIdx.x;
    const int tok = tokens[row];
    for (int c = threadIdx.x; c < DIMd; c += blockDim.x)
        x[(size_t)row * DIMd + c] = tok_emb[(size_t)tok * DIMd + c];
}

// ---------------------------------------------------------------------------
// LayerNorm over last dim (512), one block per row
// ---------------------------------------------------------------------------
__global__ void layernorm_k(const float* __restrict__ x,
                            const float* __restrict__ g,
                            const float* __restrict__ b,
                            float* __restrict__ y)
{
    const int row = blockIdx.x;
    const float* xr = x + (size_t)row * DIMd;
    __shared__ float red[256];
    float s = 0.f;
    for (int c = threadIdx.x; c < DIMd; c += blockDim.x) s += xr[c];
    red[threadIdx.x] = s; __syncthreads();
    for (int st = blockDim.x >> 1; st > 0; st >>= 1) {
        if ((int)threadIdx.x < st) red[threadIdx.x] += red[threadIdx.x + st];
        __syncthreads();
    }
    const float mean = red[0] * (1.0f / DIMd);
    __syncthreads();
    float sv = 0.f;
    for (int c = threadIdx.x; c < DIMd; c += blockDim.x) {
        float d = xr[c] - mean; sv += d * d;
    }
    red[threadIdx.x] = sv; __syncthreads();
    for (int st = blockDim.x >> 1; st > 0; st >>= 1) {
        if ((int)threadIdx.x < st) red[threadIdx.x] += red[threadIdx.x + st];
        __syncthreads();
    }
    const float rstd = rsqrtf(red[0] * (1.0f / DIMd) + 1e-5f);
    for (int c = threadIdx.x; c < DIMd; c += blockDim.x)
        y[(size_t)row * DIMd + c] = (xr[c] - mean) * rstd * g[c] + b[c];
}

// ---------------------------------------------------------------------------
// Tiled WMMA GEMM: C[M,N] = act(A[M,K] @ W[K,N] + bias) + resid
// 256 threads = 8 waves; block tile 64x128; wave tile 32x32 (2x2 WMMA tiles).
// A/W tiles staged in LDS (f32, padded rows) via double-buffered TDM DMA:
// the next K-tile's tensor_load_to_lds rides under the current tile's WMMAs.
// ---------------------------------------------------------------------------
#define BM 64
#define BN 128
#define BK 32
#define A_STRIDE 33    // 32 + 1 pad dword
#define B_STRIDE 129   // 128 + 1 pad dword

__device__ __forceinline__ void gemm_compute_step(const float* __restrict__ As,
                                                  const float* __restrict__ Bs,
                                                  int waveM, int waveN,
                                                  int half, int l16,
                                                  v8f acc[2][2])
{
    v16h aA[2], bB[2];
    #pragma unroll
    for (int mt = 0; mt < 2; ++mt) {
        const float* ar = As + (waveM * 32 + mt * 16 + l16) * A_STRIDE + 8 * half;
        #pragma unroll
        for (int j = 0; j < 8; ++j) {
            aA[mt][j]     = (_Float16)ar[j];
            aA[mt][j + 8] = (_Float16)ar[j + 16];
        }
    }
    #pragma unroll
    for (int nt = 0; nt < 2; ++nt) {
        const float* br = Bs + (8 * half) * B_STRIDE + waveN * 32 + nt * 16 + l16;
        #pragma unroll
        for (int j = 0; j < 8; ++j) {
            bB[nt][j]     = (_Float16)br[(size_t)j * B_STRIDE];
            bB[nt][j + 8] = (_Float16)br[(size_t)(j + 16) * B_STRIDE];
        }
    }
    #pragma unroll
    for (int mt = 0; mt < 2; ++mt)
        #pragma unroll
        for (int nt = 0; nt < 2; ++nt)
            acc[mt][nt] = __builtin_amdgcn_wmma_f32_16x16x32_f16(
                false, aA[mt], false, bB[nt], (short)0, acc[mt][nt], false, false);
}

__global__ __launch_bounds__(256)
void gemm_wmma_k(const float* __restrict__ A, int lda,
                 const float* __restrict__ W, int ldw,
                 const float* __restrict__ bias,
                 const float* __restrict__ resid,
                 float* __restrict__ C, int ldc,
                 int Ksize, int act)
{
    __shared__ float As[2][BM * A_STRIDE];   // 2 x 8448 B
    __shared__ float Bs[2][BK * B_STRIDE];   // 2 x 16512 B

    const int tid  = threadIdx.x;
    const int wave = tid >> 5;
    const int lane = tid & 31;
    const int half = lane >> 4;
    const int l16  = lane & 15;
    const int waveM = wave >> 2;          // 0..1
    const int waveN = wave & 3;           // 0..3
    const int blockM0 = blockIdx.y * BM;
    const int blockN0 = blockIdx.x * BN;

    v8f acc[2][2] = {};

#if HAVE_TDM
    // prologue: DMA first K-tile into buffer 0
    if (wave == 0) {
        tdm_load_2d(lds_off32(As[0]), A + (size_t)blockM0 * lda,
                    BK, BM, (unsigned long long)lda, 4, 1);
        tdm_load_2d(lds_off32(Bs[0]), W + blockN0,
                    BN, BK, (unsigned long long)ldw, 6, 1);
        __builtin_amdgcn_s_wait_tensorcnt(0);
    }
    __syncthreads();
    int buf = 0;
    for (int k0 = 0; k0 < Ksize; k0 += BK) {
        const bool more = (k0 + BK) < Ksize;
        // kick off next tile's DMA before computing this one (overlap)
        if (wave == 0 && more) {
            tdm_load_2d(lds_off32(As[buf ^ 1]), A + (size_t)blockM0 * lda + (k0 + BK),
                        BK, BM, (unsigned long long)lda, 4, 1);
            tdm_load_2d(lds_off32(Bs[buf ^ 1]), W + (size_t)(k0 + BK) * ldw + blockN0,
                        BN, BK, (unsigned long long)ldw, 6, 1);
        }
        gemm_compute_step(As[buf], Bs[buf], waveM, waveN, half, l16, acc);
        __syncthreads();                                     // all done reading buf
        if (wave == 0 && more) __builtin_amdgcn_s_wait_tensorcnt(0);
        __syncthreads();                                     // publish buf^1
        buf ^= 1;
    }
#else
    for (int k0 = 0; k0 < Ksize; k0 += BK) {
        __syncthreads();
        for (int t = tid; t < BM * BK; t += 256) {
            const int r = t >> 5, c = t & 31;
            As[0][r * A_STRIDE + c] = A[(size_t)(blockM0 + r) * lda + k0 + c];
        }
        for (int t = tid; t < BK * BN; t += 256) {
            const int r = t >> 7, c = t & 127;
            Bs[0][r * B_STRIDE + c] = W[(size_t)(k0 + r) * ldw + blockN0 + c];
        }
        __syncthreads();
        gemm_compute_step(As[0], Bs[0], waveM, waveN, half, l16, acc);
    }
#endif

    #pragma unroll
    for (int mt = 0; mt < 2; ++mt) {
        const int row0 = blockM0 + waveM * 32 + mt * 16 + half * 8;
        #pragma unroll
        for (int nt = 0; nt < 2; ++nt) {
            const int col = blockN0 + waveN * 32 + nt * 16 + l16;
            #pragma unroll
            for (int i = 0; i < 8; ++i) {
                const int r = row0 + i;
                float v = acc[mt][nt][i];
                if (bias) v += bias[col];
                if (act == 1) v = 0.5f * v * (1.0f + erff(v * 0.70710678118f));
                if (resid) v += resid[(size_t)r * ldc + col];
                C[(size_t)r * ldc + col] = v;
            }
        }
    }
}

// ---------------------------------------------------------------------------
// Fused causal self-attention with T5 relative-position bias (flash style).
// One wave per (b, h, 16-query tile). q: [B,N,H*DH]; kv: [B,N,2*DH].
// 32-key KV tiles double-buffered in LDS via TDM (in-order TENSORcnt lets a
// single wave wait to depth 1 while the next tile streams in).
// ---------------------------------------------------------------------------
#define KV_STRIDE 129   // 128 + 1 pad dword

__global__ __launch_bounds__(32)
void attn_local_k(const float* __restrict__ q,
                  const float* __restrict__ kv,
                  const float* __restrict__ rel_emb,   // [32, HEADS]
                  float* __restrict__ out)
{
    const int lane = threadIdx.x & 31;
    const int half = lane >> 4;
    const int l16  = lane & 15;
    const int qt = blockIdx.x, h = blockIdx.y, b = blockIdx.z;
    const int q0 = qt * 16;
    const int r0 = half * 8;

    __shared__ float    kvs[2][32 * KV_STRIDE];   // 2 x 32 keys x (64 K | 64 V)
    __shared__ _Float16 lds_p[16][40];            // P tile 16x32 (f16) + pad

    // Q A-fragments for dh chunks 0..31 / 32..63
    v16h aq0, aq1;
    {
        const float* qrow = q + ((size_t)(b * Nn + q0 + l16)) * (HEADSh * DHd) + h * DHd;
        #pragma unroll
        for (int j = 0; j < 8; ++j) {
            aq0[j]     = (_Float16)qrow[8 * half + j];
            aq0[j + 8] = (_Float16)qrow[8 * half + j + 16];
            aq1[j]     = (_Float16)qrow[32 + 8 * half + j];
            aq1[j + 8] = (_Float16)qrow[32 + 8 * half + j + 16];
        }
    }

    v8f o[4] = {};
    float rm[8], rl[8];
    #pragma unroll
    for (int i = 0; i < 8; ++i) { rm[i] = -1e30f; rl[i] = 0.f; }

    const float log8_inv16 = 16.0f / logf(8.0f);
    const int kbmax = (q0 + 15) >> 5;
    int buf = 0;

#if HAVE_TDM
    tdm_load_2d(lds_off32(kvs[0]), kv + ((size_t)(b * Nn)) * (2 * DHd),
                2 * DHd, 32, (unsigned long long)(2 * DHd), 6, 1);
#endif

    for (int kb = 0; kb <= kbmax; ++kb) {
#if HAVE_TDM
        if (kb < kbmax) {
            tdm_load_2d(lds_off32(kvs[buf ^ 1]),
                        kv + ((size_t)(b * Nn + (kb + 1) * 32)) * (2 * DHd),
                        2 * DHd, 32, (unsigned long long)(2 * DHd), 6, 1);
            __builtin_amdgcn_s_wait_tensorcnt(1);   // oldest (current tile) done
        } else {
            __builtin_amdgcn_s_wait_tensorcnt(0);
        }
#else
        for (int t = lane; t < 32 * 128; t += 32) {
            const int r = t >> 7, c = t & 127;
            kvs[buf][r * KV_STRIDE + c] =
                kv[((size_t)(b * Nn + kb * 32 + r)) * (2 * DHd) + c];
        }
        __syncthreads();
#endif
        const float* kvt = kvs[buf];

        float scr[2][8];
        #pragma unroll
        for (int t = 0; t < 2; ++t) {
            const int kt = kb * 32 + t * 16;
            v16h bk0, bk1;
            const float* krow = kvt + (t * 16 + l16) * KV_STRIDE;
            #pragma unroll
            for (int j = 0; j < 8; ++j) {
                bk0[j]     = (_Float16)krow[8 * half + j];
                bk0[j + 8] = (_Float16)krow[8 * half + j + 16];
                bk1[j]     = (_Float16)krow[32 + 8 * half + j];
                bk1[j + 8] = (_Float16)krow[32 + 8 * half + j + 16];
            }
            v8f c = {};
            c = __builtin_amdgcn_wmma_f32_16x16x32_f16(false, aq0, false, bk0,
                                                       (short)0, c, false, false);
            c = __builtin_amdgcn_wmma_f32_16x16x32_f16(false, aq1, false, bk1,
                                                       (short)0, c, false, false);
            const int key = kt + l16;
            #pragma unroll
            for (int i = 0; i < 8; ++i) {
                const int qi = q0 + r0 + i;
                float s = c[i] * SCALEf;
                if (key > qi) {
                    s = -1e30f;                      // causal mask
                } else {
                    const int neg = qi - key;
                    int bucket;
                    if (neg < 16) bucket = neg;
                    else {
                        int vb = 16 + (int)(logf((float)neg * 0.0625f) * log8_inv16);
                        bucket = vb < 31 ? vb : 31;
                    }
                    s += rel_emb[bucket * HEADSh + h] * SQRTDH;
                }
                scr[t][i] = s;
            }
        }
        // online softmax over this 32-key block
        #pragma unroll
        for (int i = 0; i < 8; ++i) {
            float mx = fmaxf(scr[0][i], scr[1][i]);
            #pragma unroll
            for (int d = 1; d < 16; d <<= 1) mx = fmaxf(mx, __shfl_xor(mx, d, 32));
            const float nm = fmaxf(rm[i], mx);
            const float alpha = __expf(rm[i] - nm);
            const float p0 = __expf(scr[0][i] - nm);
            const float p1 = __expf(scr[1][i] - nm);
            float rs = p0 + p1;
            #pragma unroll
            for (int d = 1; d < 16; d <<= 1) rs += __shfl_xor(rs, d, 32);
            rl[i] = rl[i] * alpha + rs;
            rm[i] = nm;
            o[0][i] *= alpha; o[1][i] *= alpha; o[2][i] *= alpha; o[3][i] *= alpha;
            lds_p[r0 + i][l16]      = (_Float16)p0;
            lds_p[r0 + i][16 + l16] = (_Float16)p1;
        }
        __syncthreads();
        // P (16x32) as A-fragment
        v16h ap;
        #pragma unroll
        for (int j = 0; j < 8; ++j) {
            ap[j]     = lds_p[l16][8 * half + j];
            ap[j + 8] = lds_p[l16][8 * half + j + 16];
        }
        // V (32x64) as B-fragments over 4 dh tiles; O += P @ V
        #pragma unroll
        for (int dt = 0; dt < 4; ++dt) {
            v16h bv;
            const float* vcol = kvt + DHd + dt * 16 + l16;
            #pragma unroll
            for (int j = 0; j < 8; ++j) {
                bv[j]     = (_Float16)vcol[(8 * half + j) * KV_STRIDE];
                bv[j + 8] = (_Float16)vcol[(8 * half + j + 16) * KV_STRIDE];
            }
            o[dt] = __builtin_amdgcn_wmma_f32_16x16x32_f16(false, ap, false, bv,
                                                           (short)0, o[dt], false, false);
        }
        __syncthreads();
        buf ^= 1;
    }

    #pragma unroll
    for (int i = 0; i < 8; ++i) {
        const float inv = 1.0f / rl[i];
        const int qi = q0 + r0 + i;
        const size_t base = ((size_t)(b * Nn + qi)) * (HEADSh * DHd) + h * DHd;
        out[base +  0 + l16] = o[0][i] * inv;
        out[base + 16 + l16] = o[1][i] * inv;
        out[base + 32 + l16] = o[2][i] * inv;
        out[base + 48 + l16] = o[3][i] * inv;
    }
}

// ---------------------------------------------------------------------------
// Memory (kNN) attention for layer 2. One 256-thread block per (b,h,n) query.
// ---------------------------------------------------------------------------
__global__ __launch_bounds__(256)
void attn_mem_k(const float* __restrict__ q,
                const float* __restrict__ mem_kv,  // [B, M, 2, DH]
                const float* __restrict__ null_k,
                const float* __restrict__ null_v,
                const float* __restrict__ gate,    // [HEADS]
                float* __restrict__ out)           // in/out [B,N,H*DH]
{
    const int n = blockIdx.x, h = blockIdx.y, b = blockIdx.z;
    const int t = threadIdx.x;

    __shared__ float qv[DHd];
    __shared__ float sc[Mm];
    __shared__ float redv[256];
    __shared__ int   redi[256];
    __shared__ int   topi[KTOP];
    __shared__ float topv[KTOP];
    __shared__ float w[KTOP + 1];

    const size_t qbase = ((size_t)(b * Nn + n)) * (HEADSh * DHd) + h * DHd;
    if (t < DHd) qv[t] = q[qbase + t];
    __syncthreads();

    for (int m = t; m < Mm; m += 256) {
        const float* kp = mem_kv + ((size_t)(b * Mm + m)) * 2 * DHd;
        float s = 0.f;
        #pragma unroll
        for (int d = 0; d < DHd; ++d) s += qv[d] * kp[d];
        sc[m] = s;
    }
    __syncthreads();

    for (int it = 0; it < KTOP; ++it) {
        float best = -1e30f; int bi = 0;
        for (int m = t; m < Mm; m += 256)
            if (sc[m] > best) { best = sc[m]; bi = m; }
        redv[t] = best; redi[t] = bi; __syncthreads();
        for (int st = 128; st > 0; st >>= 1) {
            if (t < st && redv[t + st] > redv[t]) { redv[t] = redv[t + st]; redi[t] = redi[t + st]; }
            __syncthreads();
        }
        if (t == 0) { topi[it] = redi[0]; topv[it] = redv[0]; sc[redi[0]] = -1e30f; }
        __syncthreads();
    }

    if (t == 0) {
        float s0 = 0.f;
        for (int d = 0; d < DHd; ++d) s0 += qv[d] * null_k[d];
        s0 *= SCALEf;
        float mx = s0;
        for (int j = 0; j < KTOP; ++j) mx = fmaxf(mx, topv[j] * SCALEf);
        float sum;
        w[0] = __expf(s0 - mx); sum = w[0];
        for (int j = 0; j < KTOP; ++j) { w[j + 1] = __expf(topv[j] * SCALEf - mx); sum += w[j + 1]; }
        const float inv = 1.f / sum;
        for (int j = 0; j <= KTOP; ++j) w[j] *= inv;
    }
    __syncthreads();

    if (t < DHd) {
        float acc = w[0] * null_v[t];
        for (int j = 0; j < KTOP; ++j) {
            const float* vp = mem_kv + ((size_t)(b * Mm + topi[j])) * 2 * DHd + DHd;
            acc += w[j + 1] * vp[t];
        }
        const float g = 1.f / (1.f + __expf(-gate[h]));
        const float loc = out[qbase + t];
        out[qbase + t] = loc * g + acc * (1.f - g);
    }
}

// ---------------------------------------------------------------------------
// Host-side orchestration
// ---------------------------------------------------------------------------
extern "C" void kernel_launch(void* const* d_in, const int* in_sizes, int n_in,
                              void* d_out, int out_size, void* d_ws, size_t ws_size,
                              hipStream_t stream)
{
    (void)in_sizes; (void)n_in; (void)out_size; (void)ws_size;

    const int*   tokens  = (const int*)d_in[0];
    const float* tok_emb = (const float*)d_in[1];
    const float* rel_emb = (const float*)d_in[2];

    struct Layer {
        const float *ln1_g, *ln1_b, *wq, *wkv, *wout, *bout;
        const float *ln2_g, *ln2_b, *w1, *b1, *w2, *b2;
        const float *null_k, *null_v, *gate;
    } L[4];

    int idx = 3;
    for (int i = 0; i < 4; ++i) {
        L[i].ln1_g = (const float*)d_in[idx++];
        L[i].ln1_b = (const float*)d_in[idx++];
        L[i].wq    = (const float*)d_in[idx++];
        L[i].wkv   = (const float*)d_in[idx++];
        L[i].wout  = (const float*)d_in[idx++];
        L[i].bout  = (const float*)d_in[idx++];
        L[i].ln2_g = (const float*)d_in[idx++];
        L[i].ln2_b = (const float*)d_in[idx++];
        L[i].w1    = (const float*)d_in[idx++];
        L[i].b1    = (const float*)d_in[idx++];
        L[i].w2    = (const float*)d_in[idx++];
        L[i].b2    = (const float*)d_in[idx++];
        if (i == 1) {   // MEM_LAYER == 2 (1-based)
            L[i].null_k = (const float*)d_in[idx++];
            L[i].null_v = (const float*)d_in[idx++];
            L[i].gate   = (const float*)d_in[idx++];
        } else {
            L[i].null_k = L[i].null_v = L[i].gate = nullptr;
        }
    }
    const float* lnf_g   = (const float*)d_in[idx++];
    const float* lnf_b   = (const float*)d_in[idx++];
    const float* w_logit = (const float*)d_in[idx++];
    const float* b_logit = (const float*)d_in[idx++];
    const float* mem_kv  = (const float*)d_in[idx++];

    const int R = Bb * Nn;               // 2048 rows
    float* ws  = (float*)d_ws;
    float* x   = ws;                     // R*512
    float* h   = x   + (size_t)R * DIMd; // R*512
    float* qb  = h   + (size_t)R * DIMd; // R*512
    float* kvb = qb  + (size_t)R * DIMd; // R*128
    float* ao  = kvb + (size_t)R * 128;  // R*512
    float* ffn = ao  + (size_t)R * DIMd; // R*2048

    embed_k<<<R, 128, 0, stream>>>(tokens, tok_emb, x);

    for (int li = 0; li < 4; ++li) {
        const Layer& p = L[li];
        layernorm_k<<<R, 256, 0, stream>>>(x, p.ln1_g, p.ln1_b, h);

        // q = h @ wq [512x512], kv = h @ wkv [512x128]
        gemm_wmma_k<<<dim3(DIMd / BN, R / BM), 256, 0, stream>>>(
            h, DIMd, p.wq, DIMd, nullptr, nullptr, qb, DIMd, DIMd, 0);
        gemm_wmma_k<<<dim3(128 / BN, R / BM), 256, 0, stream>>>(
            h, DIMd, p.wkv, 128, nullptr, nullptr, kvb, 128, DIMd, 0);

        attn_local_k<<<dim3(Nn / 16, HEADSh, Bb), 32, 0, stream>>>(qb, kvb, rel_emb, ao);

        if (li == 1)
            attn_mem_k<<<dim3(Nn, HEADSh, Bb), 256, 0, stream>>>(
                qb, mem_kv, p.null_k, p.null_v, p.gate, ao);

        // x = ao @ wout + bout + x
        gemm_wmma_k<<<dim3(DIMd / BN, R / BM), 256, 0, stream>>>(
            ao, DIMd, p.wout, DIMd, p.bout, x, x, DIMd, DIMd, 0);

        layernorm_k<<<R, 256, 0, stream>>>(x, p.ln2_g, p.ln2_b, h);

        // ffn = gelu(h @ w1 + b1) [512x2048]
        gemm_wmma_k<<<dim3(2048 / BN, R / BM), 256, 0, stream>>>(
            h, DIMd, p.w1, 2048, p.b1, nullptr, ffn, 2048, DIMd, 1);
        // x = ffn @ w2 + b2 + x [2048x512]
        gemm_wmma_k<<<dim3(DIMd / BN, R / BM), 256, 0, stream>>>(
            ffn, 2048, p.w2, DIMd, p.b2, x, x, DIMd, 2048, 0);
    }

    layernorm_k<<<R, 256, 0, stream>>>(x, lnf_g, lnf_b, h);

    // logits = h @ w_logit + b_logit  [512 x 32000]
    gemm_wmma_k<<<dim3(Vv / BN, R / BM), 256, 0, stream>>>(
        h, DIMd, w_logit, Vv, b_logit, nullptr, (float*)d_out, Vv, DIMd, 0);
}
